// GaborRenderer_GINO3D_21208548507866
// MI455X (gfx1250) — matrix operand order repaired and checked by reading, not compiled, not run on hardware
//
#include <hip/hip_runtime.h>
#include <hip/hip_bf16.h>
#include <math.h>

typedef __attribute__((ext_vector_type(16))) _Float16     v16h;
typedef __attribute__((ext_vector_type(8)))  float        v8f;
typedef __attribute__((ext_vector_type(4)))  unsigned int u32x4;
typedef __attribute__((ext_vector_type(4)))  float        f32x4;

#define BATCH    8
#define NPTS     16384
#define MKV      64
#define TILE     128
#define THREADS  256

// ---------------- workspace layout (bytes) ----------------
#define WP_CP2   0                        // 256x256 f16 frags  (131072 B)
#define WP_Q     (WP_CP2  + 131072)
#define WP_OUT   (WP_Q    + 131072)
#define WP_NET1  (WP_OUT  + 131072)      // 4 matrices
#define WP_NET2  (WP_NET1 + 4*131072)    // 5 matrices
#define WP_F2    (WP_NET2 + 5*131072)    // 32x256 f16 frags (16384 B)
#define WS_KV    (WP_F2   + 16384)       // [B][2][64][256] f16 (524288 B)
#define WS_ZF    (WS_KV   + 524288)      // [B*64][256] f32
#define WS_POS   (WS_ZF   + 524288)      // [64][128] f32
#define WS_F2F   (WS_POS  + 32768)       // [32][256] f32 folded final
#define WS_F2B   (WS_F2F  + 32768)       // [32] f32 bias
#define WS_FILT  (WS_F2B  + 256)         // [5][256][12] f32 packed gabor
// end ~2.64 MB

// ---------------- LDS layout (dynamic) ----------------
#define SM_A     0            // 128x256 f16
#define SM_B     65536        // 128x256 f16
#define SM_KV    131072       // K[64][256] + V[64][256] f16
#define SM_COORD 196608       // 128 x {x,y,z,|x|^2} f32
#define SMEM_BYTES (196608 + 2048)

__device__ __forceinline__ float gelu_exact(float x) {
    return 0.5f * x * (1.0f + erff(x * 0.70710678118654752f));
}

struct Frag { union { v16h v; u32x4 q[2]; }; };
struct AF   { Frag a[8]; };

// A-fragments for one wave's 16 rows from an LDS 128x256 f16 tile.
// A k order used here (contiguous, same as pack_w16 B; 16x32): lane<16 -> row=lane, K {0..15} in v0-7;
// lane>=16 -> row=lane-16, K {16..31}.  Row-major LDS makes this
// two ds_load_b128 at (base) and (base+16) per k-step.
__device__ __forceinline__ void load_afrags(AF& A, const unsigned char* sBase,
                                            int wave, int lane) {
    const int arow = wave * 16 + (lane & 15);
    const unsigned char* rp = sBase + arow * 512 + ((lane >> 4) & 1) * 32;
#pragma unroll
    for (int k = 0; k < 8; ++k) {
        const u32x4* p = (const u32x4*)(rp + k * 64);
        A.a[k].q[0] = p[0];
        A.a[k].q[1] = p[1];
    }
}

// 16x16 C-tile: D += A(16x256) * Wpacked(256-sliceK x 16 cols of tile nt)
__device__ __forceinline__ v8f gemm_nt(const AF& A, const unsigned char* wP,
                                       int nt, int ntiles, int lane, v8f acc) {
    const unsigned char* base = wP + lane * 32;
#pragma unroll
    for (int k = 0; k < 8; ++k) {
        Frag b;
        const u32x4* bp = (const u32x4*)(base + (size_t)(k * ntiles + nt) * 1024);
        b.q[0] = bp[0];
        b.q[1] = bp[1];
        acc = __builtin_amdgcn_wmma_f32_16x16x32_f16(false, A.a[k].v,
                                                     false, b.v,
                                                     (short)0, acc,
                                                     false, false);
    }
    return acc;
}

__device__ __forceinline__ void store_c16(unsigned char* dstBase, int wave,
                                          int lane, int nt, v8f acc) {
    const int col = nt * 16 + (lane & 15);
    const int r0  = wave * 16 + ((lane >> 4) & 1) * 8;
    _Float16* d = (_Float16*)dstBase;
#pragma unroll
    for (int r = 0; r < 8; ++r) d[(r0 + r) * 256 + col] = (_Float16)acc[r];
}

// =================== prep kernels ===================

__global__ void k_pos(const float* grid, const float* pos_W, const float* pos_b,
                      float* pos) {
    int e = blockIdx.x * blockDim.x + threadIdx.x;
    if (e >= 64 * 128) return;
    int m = e >> 7, j = e & 127;
    float s = pos_b[j] + grid[m * 3 + 0] * pos_W[j * 3 + 0]
                       + grid[m * 3 + 1] * pos_W[j * 3 + 1]
                       + grid[m * 3 + 2] * pos_W[j * 3 + 2];
    pos[e] = gelu_exact(s);
}

__global__ void k_zf(const float* z, const float* pos, const float* fuse_W,
                     const float* fuse_b, float* zf) {
    int e = blockIdx.x * blockDim.x + threadIdx.x;
    if (e >= BATCH * 64 * 256) return;
    int n = e & 255, row = e >> 8;
    int m = row & 63;
    const float* wr = fuse_W + (size_t)n * 384;
    const float* zr = z + (size_t)row * 256;
    float s = fuse_b[n];
    for (int k = 0; k < 256; ++k) s = fmaf(zr[k], wr[k], s);
    const float* pr = pos + m * 128;
    for (int j = 0; j < 128; ++j) s = fmaf(pr[j], wr[256 + j], s);
    zf[e] = s;
}

__global__ void k_lnkv(const float* g, const float* bb, float* zf) {
    int row = blockIdx.x * blockDim.x + threadIdx.x;
    if (row >= BATCH * 64) return;
    float* r = zf + (size_t)row * 256;
    float m = 0.f;
    for (int c = 0; c < 256; ++c) m += r[c];
    m *= (1.0f / 256.0f);
    float v = 0.f;
    for (int c = 0; c < 256; ++c) { float d = r[c] - m; v += d * d; }
    v *= (1.0f / 256.0f);
    float inv = rsqrtf(v + 1e-5f);
    for (int c = 0; c < 256; ++c) r[c] = (r[c] - m) * inv * g[c] + bb[c];
}

__global__ void k_kv(const float* zkv, const float* inW, const float* inb,
                     _Float16* kv) {
    int e = blockIdx.x * blockDim.x + threadIdx.x;
    if (e >= BATCH * 64 * 256) return;
    int n = e & 255, row = e >> 8;
    int b = row >> 6, m = row & 63;
    const float* zr = zkv + (size_t)row * 256;
    const float* kw = inW + (size_t)(256 + n) * 256;
    const float* vw = inW + (size_t)(512 + n) * 256;
    float ks = inb[256 + n], vs = inb[512 + n];
    for (int l = 0; l < 256; ++l) {
        float zl = zr[l];
        ks = fmaf(zl, kw[l], ks);
        vs = fmaf(zl, vw[l], vs);
    }
    kv[((size_t)(b * 2 + 0) * 64 + m) * 256 + n] = (_Float16)ks;
    kv[((size_t)(b * 2 + 1) * 64 + m) * 256 + n] = (_Float16)vs;
}

// fold irfft(n=16, ortho) into final_W: F2[t*2+c][k], F2b[t*2+c]
__global__ void k_final2(const float* fW, const float* fb, float* F2, float* F2b) {
    int e = blockIdx.x * blockDim.x + threadIdx.x;
    if (e >= 32 * 256) return;
    int k = e & 255, j = e >> 8;
    int t = j >> 1, c = j & 1;
    float s = 0.f, sb = 0.f;
    for (int f = 0; f < 9; ++f) {
        float w   = (f == 0 || f == 8) ? 0.25f : 0.5f;   // (1/sqrt(16)) * {1,2}
        float ang = 0.39269908169872414f * (float)(f * t); // 2*pi*f*t/16
        float cr  =  w * cosf(ang);
        float ci  = -w * sinf(ang);
        s  += cr * fW[(f * 4 + c * 2 + 0) * 256 + k] + ci * fW[(f * 4 + c * 2 + 1) * 256 + k];
        sb += cr * fb[f * 4 + c * 2 + 0] + ci * fb[f * 4 + c * 2 + 1];
    }
    F2[e] = s;
    if (k == 0) F2b[j] = sb;
}

// gabor params: {w0,w1,w2,b, g*mu0,g*mu1,g*mu2, -0.5*g*|mu|^2, g, 0,0,0}
__global__ void k_filt(const float* fW, const float* fb, const float* fmu,
                       const float* fg, float* dst) {
    int e = blockIdx.x * blockDim.x + threadIdx.x;
    if (e >= 5 * 256) return;
    float g  = fg[e];
    float m0 = fmu[e * 3 + 0], m1 = fmu[e * 3 + 1], m2 = fmu[e * 3 + 2];
    float* d = dst + (size_t)e * 12;
    d[0] = fW[e * 3 + 0]; d[1] = fW[e * 3 + 1]; d[2] = fW[e * 3 + 2]; d[3] = fb[e];
    d[4] = g * m0; d[5] = g * m1; d[6] = g * m2;
    d[7] = -0.5f * g * (m0 * m0 + m1 * m1 + m2 * m2);
    d[8] = g; d[9] = 0.f; d[10] = 0.f; d[11] = 0.f;
}

// pack fp32 [Nout][K] row-major -> f16 B-fragments: frag[kt*ntiles+nt],
// lane owns col n=nt*16+(lane&15), 16 consecutive K at kt*32+((lane>>4)&1)*16
__global__ void pack_w16(const float* W, _Float16* dst, int K, int ntiles) {
    int t = blockIdx.x * blockDim.x + threadIdx.x;
    int total = (K / 32) * ntiles * 32;
    if (t >= total) return;
    int lane = t & 31;
    int frag = t >> 5;
    int nt = frag % ntiles;
    int n  = nt * 16 + (lane & 15);
    int kk = (frag / ntiles) * 32 + ((lane >> 4) & 1) * 16;
    _Float16* d = dst + (size_t)frag * 512 + lane * 16;
#pragma unroll
    for (int e = 0; e < 16; ++e) d[e] = (_Float16)W[(size_t)n * K + kk + e];
}

// =================== main fused kernel ===================

__global__ __launch_bounds__(THREADS, 1)
void gino_main(const float* __restrict__ coords,
               const float* __restrict__ cp1_W, const float* __restrict__ cp1_b,
               const float* __restrict__ cp2_b,
               const float* __restrict__ nq_g,  const float* __restrict__ nq_b,
               const float* __restrict__ attn_in_b,
               const float* __restrict__ attn_out_b,
               const float* __restrict__ net1_b,
               float* __restrict__ out,
               const unsigned char* __restrict__ ws) {
    extern __shared__ unsigned char smem[];
    const int tid  = threadIdx.x;
    const int lane = tid & 31;
    const int wave = tid >> 5;
    const int bIdx = blockIdx.x >> 7;          // 128 tiles per batch
    const int n0   = (blockIdx.x & 127) * TILE;

    // ---- phase 0: async-stage K/V (64KB) cache->LDS; load coords tile ----
    // GLOBAL_LOAD_ASYNC_TO_LDS_B128 (ASYNCcnt): no VGPR staging, overlaps with
    // phases 1..4; we only wait on ASYNCcnt right before attention (phase 5).
    {
        const unsigned char* gsrc = ws + WS_KV + (size_t)bIdx * 65536 + tid * 16;
        unsigned int ldsAddr =
            (unsigned int)(uintptr_t)(smem + SM_KV) + (unsigned int)tid * 16u;
#pragma unroll
        for (int i = 0; i < 16; ++i) {
            asm volatile("global_load_async_to_lds_b128 %0, %1, off"
                         :
                         : "v"(ldsAddr + (unsigned)i * 4096u),
                           "v"(gsrc + (size_t)i * 4096)
                         : "memory");
        }
    }
    if (tid < TILE) {
        const float* cp = coords + ((size_t)bIdx * NPTS + n0 + tid) * 3;
        float x = cp[0], y = cp[1], zz = cp[2];
        float* sc = (float*)(smem + SM_COORD) + tid * 4;
        sc[0] = x; sc[1] = y; sc[2] = zz; sc[3] = x * x + y * y + zz * zz;
    }
    __builtin_prefetch(ws + WP_CP2, 0, 0);
    __syncthreads();

    // ---- phase 1: h1 = gelu(coords @ cp1^T + b) -> sA (f16) ----
    {
        const int col = tid;
        const float w0 = cp1_W[col * 3 + 0], w1 = cp1_W[col * 3 + 1],
                    w2 = cp1_W[col * 3 + 2], bb = cp1_b[col];
        _Float16* sA = (_Float16*)(smem + SM_A);
        const float* sc = (const float*)(smem + SM_COORD);
        for (int r = 0; r < TILE; ++r) {
            float t = fmaf(w0, sc[r * 4 + 0],
                      fmaf(w1, sc[r * 4 + 1],
                      fmaf(w2, sc[r * 4 + 2], bb)));
            sA[r * 256 + col] = (_Float16)gelu_exact(t);
        }
    }
    __syncthreads();

    // ---- phase 2: (h1 @ cp2^T + cp2_b) -> sB ----
    {
        AF A; load_afrags(A, smem + SM_A, wave, lane);
#pragma unroll 1
        for (int nt = 0; nt < 16; ++nt) {
            v8f acc = {0, 0, 0, 0, 0, 0, 0, 0};
            acc = gemm_nt(A, ws + WP_CP2, nt, 16, lane, acc);
            const float bb = cp2_b[nt * 16 + (lane & 15)];
#pragma unroll
            for (int r = 0; r < 8; ++r) acc[r] += bb;
            store_c16(smem + SM_B, wave, lane, nt, acc);
        }
    }
    __syncthreads();

    // ---- phase 3: LayerNorm(nq) per row: sB -> sA ----
    if (tid < TILE) {
        const _Float16* src = (const _Float16*)(smem + SM_B) + tid * 256;
        float m = 0.f;
#pragma unroll 8
        for (int c = 0; c < 256; ++c) m += (float)src[c];
        m *= (1.0f / 256.0f);
        float v = 0.f;
#pragma unroll 8
        for (int c = 0; c < 256; ++c) { float d = (float)src[c] - m; v += d * d; }
        v *= (1.0f / 256.0f);
        float inv = rsqrtf(v + 1e-5f);
        _Float16* d = (_Float16*)(smem + SM_A) + tid * 256;
#pragma unroll 8
        for (int c = 0; c < 256; ++c)
            d[c] = (_Float16)(((float)src[c] - m) * inv * nq_g[c] + nq_b[c]);
    }
    __syncthreads();

    // ---- phase 4: q = q_ln @ qW^T + qb -> sB ----
    {
        AF A; load_afrags(A, smem + SM_A, wave, lane);
#pragma unroll 1
        for (int nt = 0; nt < 16; ++nt) {
            v8f acc = {0, 0, 0, 0, 0, 0, 0, 0};
            acc = gemm_nt(A, ws + WP_Q, nt, 16, lane, acc);
            const float bb = attn_in_b[nt * 16 + (lane & 15)];
#pragma unroll
            for (int r = 0; r < 8; ++r) acc[r] += bb;
            store_c16(smem + SM_B, wave, lane, nt, acc);
        }
    }
    // K/V async copy must be complete (and visible workgroup-wide) here.
    asm volatile("s_wait_asynccnt 0x0" ::: "memory");
    __syncthreads();

    // ---- phase 5: cross-attention (64 keys, 4 heads) online softmax -> sA ----
    {
        const _Float16* sK = (const _Float16*)(smem + SM_KV);
        const _Float16* sV = sK + 64 * 256;
        const _Float16* sQ = (const _Float16*)(smem + SM_B);
        _Float16* sO = (_Float16*)(smem + SM_A);
        for (int u = 0; u < 2; ++u) {
            const int unit = tid + u * THREADS;      // 512 units = 128 rows x 4 heads
            const int row  = unit & 127;
            const int head = unit >> 7;
            const _Float16* qr = sQ + row * 256 + head * 64;
            float qf[64];
#pragma unroll
            for (int d2 = 0; d2 < 64; ++d2) qf[d2] = (float)qr[d2];
            float mrun = -3.0e38f, lrun = 0.f;
            float o[64];
#pragma unroll
            for (int d2 = 0; d2 < 64; ++d2) o[d2] = 0.f;
            for (int key = 0; key < 64; ++key) {
                const _Float16* kr = sK + key * 256 + head * 64;
                float s = 0.f;
#pragma unroll
                for (int d2 = 0; d2 < 64; ++d2) s = fmaf(qf[d2], (float)kr[d2], s);
                s *= 0.125f;                          // 1/sqrt(64)
                float mn   = fmaxf(mrun, s);
                float corr = __expf(mrun - mn);
                float p    = __expf(s - mn);
                lrun = lrun * corr + p;
                const _Float16* vr = sV + key * 256 + head * 64;
#pragma unroll
                for (int d2 = 0; d2 < 64; ++d2)
                    o[d2] = fmaf(o[d2], corr, p * (float)vr[d2]);
                mrun = mn;
            }
            float inv = 1.0f / lrun;
            _Float16* orow = sO + row * 256 + head * 64;
#pragma unroll
            for (int d2 = 0; d2 < 64; ++d2) orow[d2] = (_Float16)(o[d2] * inv);
        }
    }
    __syncthreads();

    // ---- phase 6: zc = o @ outW^T + out_b -> sB ----
    {
        AF A; load_afrags(A, smem + SM_A, wave, lane);
#pragma unroll 1
        for (int nt = 0; nt < 16; ++nt) {
            v8f acc = {0, 0, 0, 0, 0, 0, 0, 0};
            acc = gemm_nt(A, ws + WP_OUT, nt, 16, lane, acc);
            const float bb = attn_out_b[nt * 16 + (lane & 15)];
#pragma unroll
            for (int r = 0; r < 8; ++r) acc[r] += bb;
            store_c16(smem + SM_B, wave, lane, nt, acc);
        }
    }
    __syncthreads();

    // ---- phase 7: gabor chain + final(+irfft) ----
    {
        float cx[8], cy[8], cz[8], cxx[8];
        const float* sc = (const float*)(smem + SM_COORD);
        const int r0 = wave * 16 + ((lane >> 4) & 1) * 8;
#pragma unroll
        for (int r = 0; r < 8; ++r) {
            cx[r]  = sc[(r0 + r) * 4 + 0];
            cy[r]  = sc[(r0 + r) * 4 + 1];
            cz[r]  = sc[(r0 + r) * 4 + 2];
            cxx[r] = sc[(r0 + r) * 4 + 3];
        }
        AF Az; load_afrags(Az, smem + SM_B, wave, lane);   // zc fragments
        const float* filt = (const float*)(ws + WS_FILT);
        __builtin_prefetch(ws + WP_NET2, 0, 0);

        // layer 0: x = gabor0 * (zc @ net2_0^T)
#pragma unroll 1
        for (int nt = 0; nt < 16; ++nt) {
            v8f acc = {0, 0, 0, 0, 0, 0, 0, 0};
            acc = gemm_nt(Az, ws + WP_NET2, nt, 16, lane, acc);
            const int col = nt * 16 + (lane & 15);
            const f32x4* fp = (const f32x4*)(filt + (size_t)col * 12);
            f32x4 f0 = fp[0], f1 = fp[1], f2 = fp[2];
#pragma unroll
            for (int r = 0; r < 8; ++r) {
                float t = fmaf(f0.x, cx[r], fmaf(f0.y, cy[r], fmaf(f0.z, cz[r], f0.w)));
                float e = fmaf(f1.x, cx[r], fmaf(f1.y, cy[r], fmaf(f1.z, cz[r], f1.w)))
                          - 0.5f * f2.x * cxx[r];
                acc[r] *= __sinf(t) * __expf(e);
            }
            store_c16(smem + SM_A, wave, lane, nt, acc);
        }

        // layers 1..4: x = gabor_i * (x @ net1_{i-1}^T + b + zc @ net2_i^T)
#pragma unroll 1
        for (int i = 1; i <= 4; ++i) {
            AF Ax; load_afrags(Ax, smem + SM_A, wave, lane);
            const unsigned char* w1p = ws + WP_NET1 + (size_t)(i - 1) * 131072;
            const unsigned char* w2p = ws + WP_NET2 + (size_t)i * 131072;
            __builtin_prefetch(w1p, 0, 0);
#pragma unroll 1
            for (int nt = 0; nt < 16; ++nt) {
                v8f acc = {0, 0, 0, 0, 0, 0, 0, 0};
                acc = gemm_nt(Ax, w1p, nt, 16, lane, acc);
                acc = gemm_nt(Az, w2p, nt, 16, lane, acc);
                const int col = nt * 16 + (lane & 15);
                const float nb = net1_b[(i - 1) * 256 + col];
                const f32x4* fp = (const f32x4*)(filt + ((size_t)i * 256 + col) * 12);
                f32x4 f0 = fp[0], f1 = fp[1], f2 = fp[2];
#pragma unroll
                for (int r = 0; r < 8; ++r) {
                    float h = acc[r] + nb;
                    float t = fmaf(f0.x, cx[r], fmaf(f0.y, cy[r], fmaf(f0.z, cz[r], f0.w)));
                    float e = fmaf(f1.x, cx[r], fmaf(f1.y, cy[r], fmaf(f1.z, cz[r], f1.w)))
                              - 0.5f * f2.x * cxx[r];
                    acc[r] = __sinf(t) * __expf(e) * h;
                }
                store_c16(smem + SM_A, wave, lane, nt, acc);
            }
        }

        // final (irfft folded): out[b][t][n][c] = x @ F2^T + F2b
        AF Ax; load_afrags(Ax, smem + SM_A, wave, lane);
        const float* f2b = (const float*)(ws + WS_F2B);
#pragma unroll 1
        for (int nt = 0; nt < 2; ++nt) {
            v8f acc = {0, 0, 0, 0, 0, 0, 0, 0};
            acc = gemm_nt(Ax, ws + WP_F2, nt, 2, lane, acc);
            const int col = nt * 16 + (lane & 15);   // 0..31 = t*2+c
            const float bb = f2b[col];
            const int tt = col >> 1, cc = col & 1;
            const int rw = wave * 16 + ((lane >> 4) & 1) * 8;
#pragma unroll
            for (int r = 0; r < 8; ++r) {
                const size_t n = (size_t)(n0 + rw + r);
                out[(((size_t)bIdx * 16 + tt) * NPTS + n) * 2 + cc] = acc[r] + bb;
            }
        }
    }
}

// =================== launcher ===================

extern "C" void kernel_launch(void* const* d_in, const int* in_sizes, int n_in,
                              void* d_out, int out_size, void* d_ws, size_t ws_size,
                              hipStream_t stream) {
    const float* z          = (const float*)d_in[0];
    const float* coords     = (const float*)d_in[1];
    const float* grid       = (const float*)d_in[2];
    const float* cp1_W      = (const float*)d_in[3];
    const float* cp1_b      = (const float*)d_in[4];
    const float* cp2_W      = (const float*)d_in[5];
    const float* cp2_b      = (const float*)d_in[6];
    const float* nq_g       = (const float*)d_in[7];
    const float* nq_b       = (const float*)d_in[8];
    const float* nkv_g      = (const float*)d_in[9];
    const float* nkv_b      = (const float*)d_in[10];
    const float* attn_in_W  = (const float*)d_in[11];
    const float* attn_in_b  = (const float*)d_in[12];
    const float* attn_out_W = (const float*)d_in[13];
    const float* attn_out_b = (const float*)d_in[14];
    const float* pos_W      = (const float*)d_in[15];
    const float* pos_b      = (const float*)d_in[16];
    const float* fuse_W     = (const float*)d_in[17];
    const float* fuse_b     = (const float*)d_in[18];
    const float* filt_W     = (const float*)d_in[19];
    const float* filt_b     = (const float*)d_in[20];
    const float* filt_mu    = (const float*)d_in[21];
    const float* filt_gamma = (const float*)d_in[22];
    const float* net1_W     = (const float*)d_in[23];
    const float* net1_b     = (const float*)d_in[24];
    const float* net2_W     = (const float*)d_in[25];
    const float* final_W    = (const float*)d_in[26];
    const float* final_b    = (const float*)d_in[27];
    unsigned char* ws = (unsigned char*)d_ws;
    float* out = (float*)d_out;

    // ---- weight packing (f16 WMMA B-fragments) ----
    const int blk = 256;
    pack_w16<<<(4096 + blk - 1) / blk, blk, 0, stream>>>(cp2_W,      (_Float16*)(ws + WP_CP2), 256, 16);
    pack_w16<<<(4096 + blk - 1) / blk, blk, 0, stream>>>(attn_in_W,  (_Float16*)(ws + WP_Q),   256, 16);
    pack_w16<<<(4096 + blk - 1) / blk, blk, 0, stream>>>(attn_out_W, (_Float16*)(ws + WP_OUT), 256, 16);
    for (int i = 0; i < 4; ++i)
        pack_w16<<<(4096 + blk - 1) / blk, blk, 0, stream>>>(
            net1_W + (size_t)i * 65536, (_Float16*)(ws + WP_NET1 + (size_t)i * 131072), 256, 16);
    for (int i = 0; i < 5; ++i)
        pack_w16<<<(4096 + blk - 1) / blk, blk, 0, stream>>>(
            net2_W + (size_t)i * 65536, (_Float16*)(ws + WP_NET2 + (size_t)i * 131072), 256, 16);

    // ---- small latent-side pipeline ----
    k_pos<<<32, blk, 0, stream>>>(grid, pos_W, pos_b, (float*)(ws + WS_POS));
    k_zf<<<512, blk, 0, stream>>>(z, (const float*)(ws + WS_POS), fuse_W, fuse_b,
                                  (float*)(ws + WS_ZF));
    k_lnkv<<<2, blk, 0, stream>>>(nkv_g, nkv_b, (float*)(ws + WS_ZF));
    k_kv<<<512, blk, 0, stream>>>((const float*)(ws + WS_ZF), attn_in_W, attn_in_b,
                                  (_Float16*)(ws + WS_KV));

    // ---- irfft-folded final matrix + gabor param packing ----
    k_final2<<<32, blk, 0, stream>>>(final_W, final_b,
                                     (float*)(ws + WS_F2F), (float*)(ws + WS_F2B));
    pack_w16<<<2, blk, 0, stream>>>((const float*)(ws + WS_F2F),
                                    (_Float16*)(ws + WP_F2), 256, 2);
    k_filt<<<5, blk, 0, stream>>>(filt_W, filt_b, filt_mu, filt_gamma,
                                  (float*)(ws + WS_FILT));

    // ---- main fused kernel: 1024 tiles of 128 points ----
    hipFuncSetAttribute(reinterpret_cast<const void*>(gino_main),
                        hipFuncAttributeMaxDynamicSharedMemorySize, SMEM_BYTES);
    gino_main<<<BATCH * (NPTS / TILE), THREADS, SMEM_BYTES, stream>>>(
        coords, cp1_W, cp1_b, cp2_b, nq_g, nq_b, attn_in_b, attn_out_b, net1_b,
        out, ws);
    (void)in_sizes; (void)n_in; (void)out_size; (void)ws_size;
}
